// AudioEncoder_35296041239272
// MI455X (gfx1250) — matrix operand order repaired and checked by reading, not compiled, not run on hardware
//
#include <hip/hip_runtime.h>

// ---------------------------------------------------------------------------
// Types for gfx1250 WMMA (wave32)
// ---------------------------------------------------------------------------
typedef __bf16 bf16_t;
typedef __bf16 bf16x16 __attribute__((ext_vector_type(16)));
typedef float  v8f     __attribute__((ext_vector_type(8)));

struct __align__(16) U128 { unsigned int v[4]; };
struct __align__(16) Frag32 { U128 lo, hi; };

static __device__ __forceinline__ bf16_t f2bf(float f) {
  unsigned u = __builtin_bit_cast(unsigned, f);
  u = (u + 0x7FFFu + ((u >> 16) & 1u)) >> 16;
  unsigned short s = (unsigned short)u;
  return __builtin_bit_cast(bf16_t, s);
}

static __device__ __forceinline__ bf16x16 ldfrag(const bf16_t* lo, const bf16_t* hi) {
  Frag32 f;
  f.lo = *(const U128*)lo;
  f.hi = *(const U128*)hi;
  return __builtin_bit_cast(bf16x16, f);
}

// ---------------------------------------------------------------------------
// CDNA5 async global->LDS copy path (ASYNCcnt tracked), guarded so the build
// cannot regress if the toolchain lacks the builtins.
// Probe round 2: builtin exists; param0 is v4i32* in AS(1) ("__device__"),
// param1 is the LDS destination, then (i32 offset, i32 cpol).
// ---------------------------------------------------------------------------
#if defined(__has_builtin)
#if __has_builtin(__builtin_amdgcn_global_load_async_to_lds_b128) && \
    __has_builtin(__builtin_amdgcn_s_wait_asynccnt)
#define HAVE_ASYNC_LDS 1
#endif
#endif
#ifndef HAVE_ASYNC_LDS
#define HAVE_ASYNC_LDS 0
#endif

#if HAVE_ASYNC_LDS
typedef int v4i_t __attribute__((vector_size(16)));
typedef __attribute__((address_space(1))) v4i_t* GlobV4;
typedef __attribute__((address_space(3))) v4i_t* LdsV4;
static __device__ __forceinline__ void async_b128(const bf16_t* g, bf16_t* l) {
  __builtin_amdgcn_global_load_async_to_lds_b128((GlobV4)(void*)g, (LdsV4)(void*)l, 0, 0);
}
#endif

// ---------------------------------------------------------------------------
// Problem constants
// ---------------------------------------------------------------------------
// B=8 T=1024 D=1024 H=16 HD=64 KVH=4 R=256 FF=4096 K=31, M = B*T = 8192

// ---------------------------------------------------------------------------
// LayerNorm: f32 [rows, ldin] -> bf16 (or f32) [rows, ldout]
// ---------------------------------------------------------------------------
__global__ __launch_bounds__(256)
void ln_kernel(const float* __restrict__ in, int ldin, int cols,
               const float* __restrict__ gamma, const float* __restrict__ beta,
               bf16_t* __restrict__ outB, float* __restrict__ outF, int ldout)
{
  const int row = blockIdx.x;
  const float* x = in + (size_t)row * ldin;
  float s = 0.f, s2 = 0.f;
  for (int c = threadIdx.x; c < cols; c += 256) { float v = x[c]; s += v; s2 += v * v; }
  for (int m = 16; m >= 1; m >>= 1) { s += __shfl_xor(s, m, 32); s2 += __shfl_xor(s2, m, 32); }
  __shared__ float red[16];
  const int wave = threadIdx.x >> 5, lane = threadIdx.x & 31;
  if (lane == 0) { red[wave] = s; red[wave + 8] = s2; }
  __syncthreads();
  if (wave == 0) {
    float a = (lane < 8) ? red[lane] : 0.f;
    float b = (lane < 8) ? red[lane + 8] : 0.f;
    for (int m = 4; m >= 1; m >>= 1) { a += __shfl_xor(a, m, 32); b += __shfl_xor(b, m, 32); }
    if (lane == 0) { red[0] = a; red[1] = b; }
  }
  __syncthreads();
  const float inv = 1.f / (float)cols;
  float mean = red[0] * inv;
  float var  = red[1] * inv - mean * mean;
  float rstd = rsqrtf(var + 1e-5f);
  for (int c = threadIdx.x; c < cols; c += 256) {
    float y = (x[c] - mean) * rstd * gamma[c] + beta[c];
    if (outB) outB[(size_t)row * ldout + c] = f2bf(y);
    else      outF[(size_t)row * ldout + c] = y;
  }
}

// ---------------------------------------------------------------------------
// Weight convert+transpose: f32 W[K,N] -> bf16 Wt[Npad,K]  (zero pad n>=N)
// ---------------------------------------------------------------------------
__global__ __launch_bounds__(256)
void convert_wt(const float* __restrict__ W, int K, int N,
                bf16_t* __restrict__ Wt, int total)
{
  int idx = blockIdx.x * 256 + threadIdx.x;
  if (idx >= total) return;
  int n = idx / K, k = idx - n * K;
  float v = (n < N) ? W[(size_t)k * N + n] : 0.f;
  Wt[idx] = f2bf(v);
}

// ---------------------------------------------------------------------------
// WMMA GEMM: out = epi(A[M,K](bf16) @ Wt[N,K]^T(bf16) + bias)
//   EPI 0: outF = acc+bias
//   EPI 1: outF = res + scale*(acc+bias)
//   EPI 2: outB = silu(acc+bias)  (bf16)
// Tile: 128x128x32, 256 threads = 8 waves (2 x 4), wave does 64x32 (4x2 tiles)
// Double-buffered async global->LDS pipeline when available.
// ---------------------------------------------------------------------------
template<int EPI>
__global__ __launch_bounds__(256)
void gemm_wmma(const bf16_t* __restrict__ A, int lda,
               const bf16_t* __restrict__ Wt, int ldw,
               const float* __restrict__ bias,
               const float* __restrict__ res, float scale,
               float* __restrict__ outF, bf16_t* __restrict__ outB, int ldc,
               int Kdim)
{
#if HAVE_ASYNC_LDS
  __shared__ __align__(16) bf16_t As[2][128 * 40];
  __shared__ __align__(16) bf16_t Bs[2][128 * 40];
#else
  __shared__ __align__(16) bf16_t As[1][128 * 40];
  __shared__ __align__(16) bf16_t Bs[1][128 * 40];
#endif
  const int tid  = threadIdx.x;
  const int lane = tid & 31;
  const int wave = tid >> 5;
  const int wm = wave >> 2;           // 0..1
  const int wn = wave & 3;            // 0..3
  const int g  = lane >> 4;           // lane group 0/1
  const int ln = lane & 15;
  const int m0 = blockIdx.y * 128;
  const int n0 = blockIdx.x * 128;
  const int lrow = tid >> 1;          // 0..127
  const int lcol = (tid & 1) << 4;    // 0 / 16

  const v8f vzero = {0.f,0.f,0.f,0.f,0.f,0.f,0.f,0.f};
  v8f acc[4][2];
  #pragma unroll
  for (int i = 0; i < 4; i++)
    #pragma unroll
    for (int j = 0; j < 2; j++) acc[i][j] = vzero;

  const bf16_t* aptr = A  + (size_t)(m0 + lrow) * lda + lcol;
  const bf16_t* bptr = Wt + (size_t)(n0 + lrow) * ldw + lcol;

#if HAVE_ASYNC_LDS
  // Prologue: async-issue tile 0 into buffer 0 (4 x b128 per thread).
  async_b128(aptr,     &As[0][lrow * 40 + lcol]);
  async_b128(aptr + 8, &As[0][lrow * 40 + lcol + 8]);
  async_b128(bptr,     &Bs[0][lrow * 40 + lcol]);
  async_b128(bptr + 8, &Bs[0][lrow * 40 + lcol + 8]);
#endif

  for (int k0 = 0; k0 < Kdim; k0 += 32) {
#if HAVE_ASYNC_LDS
    const int buf = (k0 >> 5) & 1;
    __builtin_amdgcn_s_wait_asynccnt(0);   // this wave's tile-k0 loads landed
    __syncthreads();                       // all waves' tiles landed; prev compute done
    if (k0 + 32 < Kdim) {                  // prefetch next tile into other buffer
      async_b128(aptr + k0 + 32,     &As[buf ^ 1][lrow * 40 + lcol]);
      async_b128(aptr + k0 + 40,     &As[buf ^ 1][lrow * 40 + lcol + 8]);
      async_b128(bptr + k0 + 32,     &Bs[buf ^ 1][lrow * 40 + lcol]);
      async_b128(bptr + k0 + 40,     &Bs[buf ^ 1][lrow * 40 + lcol + 8]);
    }
#else
    const int buf = 0;
    U128 a0 = *(const U128*)(aptr + k0);
    U128 a1 = *(const U128*)(aptr + k0 + 8);
    U128 b0 = *(const U128*)(bptr + k0);
    U128 b1 = *(const U128*)(bptr + k0 + 8);
    __builtin_prefetch(aptr + k0 + 32, 0, 1);
    __builtin_prefetch(bptr + k0 + 32, 0, 1);
    *(U128*)&As[0][lrow * 40 + lcol]     = a0;
    *(U128*)&As[0][lrow * 40 + lcol + 8] = a1;
    *(U128*)&Bs[0][lrow * 40 + lcol]     = b0;
    *(U128*)&Bs[0][lrow * 40 + lcol + 8] = b1;
    __syncthreads();
#endif

    bf16x16 af[4], bfrag[2];
    #pragma unroll
    for (int mt = 0; mt < 4; mt++) {
      const bf16_t* p = &As[buf][(wm * 64 + mt * 16 + ln) * 40 + 16 * g];
      af[mt] = ldfrag(p, p + 8);                  // K = 16g..16g+15
    }
    #pragma unroll
    for (int nt = 0; nt < 2; nt++) {
      const bf16_t* p = &Bs[buf][(wn * 32 + nt * 16 + ln) * 40 + 16 * g];
      bfrag[nt] = ldfrag(p, p + 8);               // K = 16g..16g+15
    }
    #pragma unroll
    for (int mt = 0; mt < 4; mt++)
      #pragma unroll
      for (int nt = 0; nt < 2; nt++)
        acc[mt][nt] = __builtin_amdgcn_wmma_f32_16x16x32_bf16(
            false, af[mt], false, bfrag[nt], (short)0, acc[mt][nt], false, false);
#if !HAVE_ASYNC_LDS
    __syncthreads();
#endif
  }

  #pragma unroll
  for (int mt = 0; mt < 4; mt++) {
    #pragma unroll
    for (int nt = 0; nt < 2; nt++) {
      const int gcol = n0 + wn * 32 + nt * 16 + ln;
      const float bv = bias ? bias[gcol] : 0.f;
      #pragma unroll
      for (int r = 0; r < 8; r++) {
        const int grow = m0 + wm * 64 + mt * 16 + r + 8 * g;
        float v = acc[mt][nt][r] + bv;
        const size_t o = (size_t)grow * ldc + gcol;
        if (EPI == 2)      { float sg = 1.f / (1.f + __expf(-v)); outB[o] = f2bf(v * sg); }
        else if (EPI == 1) { outF[o] = res[o] + scale * v; }
        else               { outF[o] = v; }
      }
    }
  }
}

// ---------------------------------------------------------------------------
// RoPE + head pack: f32 [B*T, ld] (cols col_off + h*64 + d) -> bf16 [B,nh,T,64]
// ---------------------------------------------------------------------------
__global__ __launch_bounds__(256)
void rope_pack(const float* __restrict__ src, int ld, int col_off, int nh,
               bf16_t* __restrict__ dst, int do_rope, int total)
{
  int idx = blockIdx.x * 256 + threadIdx.x;
  if (idx >= total) return;
  int d = idx & 63;
  int rest = idx >> 6;
  int t = rest & 1023;
  rest >>= 10;
  int h = rest % nh;
  int b = rest / nh;
  const float* row = src + (size_t)(b * 1024 + t) * ld + col_off + h * 64;
  float v = row[d];
  if (do_rope) {
    int i = d & 31;
    float ang = (float)t * __powf(10000.f, -(float)i * (1.f / 32.f));
    float c = __cosf(ang), s = __sinf(ang);
    float other = (d < 32) ? -row[d + 32] : row[d - 32];
    v = v * c + other * s;
  }
  dst[idx] = f2bf(v);
}

// V transpose pack: f32 kv [B*T, 512] cols 256.. -> bf16 vT [B,KVH,64,T]
__global__ __launch_bounds__(256)
void pack_vT(const float* __restrict__ kvf, bf16_t* __restrict__ vT, int total)
{
  int idx = blockIdx.x * 256 + threadIdx.x;
  if (idx >= total) return;
  int t = idx & 1023;
  int d = (idx >> 10) & 63;
  int h = (idx >> 16) & 3;
  int b = idx >> 18;
  vT[idx] = f2bf(kvf[(size_t)(b * 1024 + t) * 512 + 256 + h * 64 + d]);
}

// ---------------------------------------------------------------------------
// Flash attention (non-causal, full softmax). One wave = 16 query rows.
// q [B,16,T,64] bf16; k [B,4,T,64] bf16; vT [B,4,64,T] bf16; o [B*T, 1024] bf16
// ---------------------------------------------------------------------------
__global__ __launch_bounds__(256)
void flash_attn(const bf16_t* __restrict__ q, const bf16_t* __restrict__ k,
                const bf16_t* __restrict__ vT, bf16_t* __restrict__ o)
{
  __shared__ __align__(16) bf16_t pshare[8][16 * 32];
  const int lane = threadIdx.x & 31;
  const int wave = threadIdx.x >> 5;
  const int g = lane >> 4, ln = lane & 15;
  const int bidx = blockIdx.x;
  const int qb = bidx & 7;             // T/128 = 8 q-blocks
  const int h  = (bidx >> 3) & 15;
  const int b  = bidx >> 7;
  const int q0 = qb * 128 + wave * 16;
  const bf16_t* qbase = q  + (((size_t)(b * 16 + h)) * 1024 + q0) * 64;
  const bf16_t* kbase = k  + ((size_t)(b * 4 + (h >> 2))) * 1024 * 64;
  const bf16_t* vbase = vT + ((size_t)(b * 4 + (h >> 2))) * 64 * 1024;

  const bf16_t* qp = qbase + (size_t)ln * 64;        // row m = ln
  const bf16x16 aq0 = ldfrag(qp + 16 * g,      qp + 16 * g + 8);   // d 0..31
  const bf16x16 aq1 = ldfrag(qp + 32 + 16 * g, qp + 40 + 16 * g);  // d 32..63

  const v8f vzero = {0.f,0.f,0.f,0.f,0.f,0.f,0.f,0.f};
  v8f oacc[4] = {vzero, vzero, vzero, vzero};
  float m_i[8], l_i[8];
  #pragma unroll
  for (int r = 0; r < 8; r++) { m_i[r] = -3.0e38f; l_i[r] = 0.f; }

  bf16_t* pw = pshare[wave];

  for (int nk0 = 0; nk0 < 1024; nk0 += 32) {
    v8f s[2];
    #pragma unroll
    for (int nt = 0; nt < 2; nt++) {
      const bf16_t* kp = kbase + (size_t)(nk0 + nt * 16 + ln) * 64;  // col n = K row
      bf16x16 bk0 = ldfrag(kp + 16 * g,      kp + 16 * g + 8);       // d 0..31
      bf16x16 bk1 = ldfrag(kp + 32 + 16 * g, kp + 32 + 16 * g + 8);  // d 32..63
      v8f st = vzero;
      st = __builtin_amdgcn_wmma_f32_16x16x32_bf16(false, aq0, false, bk0, (short)0, st, false, false);
      st = __builtin_amdgcn_wmma_f32_16x16x32_bf16(false, aq1, false, bk1, (short)0, st, false, false);
      s[nt] = st;
    }
    float alpha[8];
    #pragma unroll
    for (int r = 0; r < 8; r++) {
      float s0 = s[0][r] * 0.125f;   // HD^-0.5
      float s1 = s[1][r] * 0.125f;
      float mx = fmaxf(s0, s1);
      mx = fmaxf(mx, __shfl_xor(mx, 1, 32));
      mx = fmaxf(mx, __shfl_xor(mx, 2, 32));
      mx = fmaxf(mx, __shfl_xor(mx, 4, 32));
      mx = fmaxf(mx, __shfl_xor(mx, 8, 32));
      float mnew = fmaxf(m_i[r], mx);
      float a  = __expf(m_i[r] - mnew);
      float p0 = __expf(s0 - mnew);
      float p1 = __expf(s1 - mnew);
      float ps = p0 + p1;
      ps += __shfl_xor(ps, 1, 32);
      ps += __shfl_xor(ps, 2, 32);
      ps += __shfl_xor(ps, 4, 32);
      ps += __shfl_xor(ps, 8, 32);
      l_i[r] = l_i[r] * a + ps;
      m_i[r] = mnew;
      alpha[r] = a;
      s[0][r] = p0;
      s[1][r] = p1;
    }
    #pragma unroll
    for (int dt = 0; dt < 4; dt++)
      #pragma unroll
      for (int r = 0; r < 8; r++) oacc[dt][r] *= alpha[r];

    // C-layout (m = r+8g, n = ln) -> LDS row-major [16][32] (per-wave private)
    #pragma unroll
    for (int r = 0; r < 8; r++) {
      pw[(r + 8 * g) * 32 + ln]      = f2bf(s[0][r]);
      pw[(r + 8 * g) * 32 + 16 + ln] = f2bf(s[1][r]);
    }
    const bf16_t* pr = pw + ln * 32;
    bf16x16 ap = ldfrag(pr + 16 * g, pr + 16 * g + 8);  // A-layout P fragment
    #pragma unroll
    for (int dt = 0; dt < 4; dt++) {
      const bf16_t* vp = vbase + (size_t)(dt * 16 + ln) * 1024 + nk0 + 16 * g;
      bf16x16 bv = ldfrag(vp, vp + 8);
      oacc[dt] = __builtin_amdgcn_wmma_f32_16x16x32_bf16(
          false, ap, false, bv, (short)0, oacc[dt], false, false);
    }
  }
  #pragma unroll
  for (int dt = 0; dt < 4; dt++)
    #pragma unroll
    for (int r = 0; r < 8; r++) {
      const int trow = q0 + r + 8 * g;
      const size_t off = ((size_t)(b * 1024 + trow)) * 1024 + h * 64 + dt * 16 + ln;
      o[off] = f2bf(oacc[dt][r] / l_i[r]);
    }
}

// ---------------------------------------------------------------------------
// GLU: f32 [M,2048] -> f32 [M,1024]:  a * sigmoid(gate)
// ---------------------------------------------------------------------------
__global__ __launch_bounds__(256)
void glu_kernel(const float* __restrict__ in, float* __restrict__ out, int total)
{
  int idx = blockIdx.x * 256 + threadIdx.x;
  if (idx >= total) return;
  int m = idx >> 10, d = idx & 1023;
  float a  = in[(size_t)m * 2048 + d];
  float gt = in[(size_t)m * 2048 + 1024 + d];
  out[idx] = a / (1.f + __expf(-gt));
}

// ---------------------------------------------------------------------------
// Depthwise conv(K=31, same pad) + BN(running stats) + SiLU -> bf16 [M, D]
// ---------------------------------------------------------------------------
__global__ __launch_bounds__(256)
void dwconv_bn_silu(const float* __restrict__ gin, const float* __restrict__ w,
                    const float* __restrict__ wb, const float* __restrict__ bng,
                    const float* __restrict__ bnb, const float* __restrict__ rm,
                    const float* __restrict__ rv, bf16_t* __restrict__ out, int total)
{
  int idx = blockIdx.x * 256 + threadIdx.x;
  if (idx >= total) return;
  int d = idx & 1023;
  int t = (idx >> 10) & 1023;
  int b = idx >> 20;
  float acc = wb[d];
  #pragma unroll
  for (int j = 0; j < 31; j++) {
    int tt = t + j - 15;
    if (tt >= 0 && tt < 1024)
      acc += gin[((size_t)(b * 1024 + tt)) * 1024 + d] * w[d * 31 + j];
  }
  float v = (acc - rm[d]) * rsqrtf(rv[d] + 1e-5f) * bng[d] + bnb[d];
  out[idx] = f2bf(v / (1.f + __expf(-v)));   // silu(v) = v * sigmoid(v)
}

// ---------------------------------------------------------------------------
// Workspace layout
// ---------------------------------------------------------------------------
constexpr size_t SZ_A   = 8192ull * 1024 * 2;
constexpr size_t SZ_MID = 8192ull * 4096 * 2;
constexpr size_t SZ_X   = 8192ull * 1024 * 4;
constexpr size_t SZ_TMP = 8192ull * 2048 * 4;
constexpr size_t SZ_LAT = 8192ull * 256 * 2;
constexpr size_t SZ_Q   = 8ull * 16 * 1024 * 64 * 2;
constexpr size_t SZ_K   = 8ull * 4 * 1024 * 64 * 2;
constexpr size_t SZ_VT  = SZ_K;
constexpr size_t SZ_GLU = 8192ull * 1024 * 4;

constexpr size_t OFF_A   = 0;
constexpr size_t OFF_MID = OFF_A + SZ_A;
constexpr size_t OFF_X   = OFF_MID + SZ_MID;
constexpr size_t OFF_TMP = OFF_X + SZ_X;
constexpr size_t OFF_LAT = OFF_TMP + SZ_TMP;
constexpr size_t OFF_Q   = OFF_LAT + SZ_LAT;
constexpr size_t OFF_K   = OFF_Q + SZ_Q;
constexpr size_t OFF_VT  = OFF_K + SZ_K;
constexpr size_t OFF_GLU = OFF_VT + SZ_VT;
constexpr size_t OFF_W0  = OFF_GLU + SZ_GLU;             // ff1_w1t [4096,1024]
constexpr size_t OFF_W1  = OFF_W0 + 4096ull * 1024 * 2;  // ff1_w2t [1024,4096]
constexpr size_t OFF_W2  = OFF_W1 + 1024ull * 4096 * 2;  // wqt     [1024,1024]
constexpr size_t OFF_W3  = OFF_W2 + 1024ull * 1024 * 2;  // wkvat   [384,1024] (pad)
constexpr size_t OFF_W4  = OFF_W3 + 384ull  * 1024 * 2;  // wkvbt   [512,256]
constexpr size_t OFF_W5  = OFF_W4 + 512ull  * 256  * 2;  // wot     [1024,1024]
constexpr size_t OFF_W6  = OFF_W5 + 1024ull * 1024 * 2;  // pw1t    [2048,1024]
constexpr size_t OFF_W7  = OFF_W6 + 2048ull * 1024 * 2;  // pw2t    [1024,1024]
constexpr size_t OFF_W8  = OFF_W7 + 1024ull * 1024 * 2;  // ff2_w1t [4096,1024]
constexpr size_t OFF_W9  = OFF_W8 + 4096ull * 1024 * 2;  // ff2_w2t [1024,4096]

static void launch_gemm(int epi, const bf16_t* A, int lda, const bf16_t* Wt, int ldw,
                        const float* bias, const float* res, float scale,
                        float* outF, bf16_t* outB, int ldc,
                        int Mdim, int Ndim, int Kdim, hipStream_t stream)
{
  dim3 grid(Ndim / 128, Mdim / 128), block(256);
  switch (epi) {
    case 0: gemm_wmma<0><<<grid, block, 0, stream>>>(A, lda, Wt, ldw, bias, res, scale, outF, outB, ldc, Kdim); break;
    case 1: gemm_wmma<1><<<grid, block, 0, stream>>>(A, lda, Wt, ldw, bias, res, scale, outF, outB, ldc, Kdim); break;
    default: gemm_wmma<2><<<grid, block, 0, stream>>>(A, lda, Wt, ldw, bias, res, scale, outF, outB, ldc, Kdim); break;
  }
}

extern "C" void kernel_launch(void* const* d_in, const int* in_sizes, int n_in,
                              void* d_out, int out_size, void* d_ws, size_t ws_size,
                              hipStream_t stream)
{
  (void)in_sizes; (void)n_in; (void)out_size; (void)ws_size;
  const float* x       = (const float*)d_in[0];
  const float* ff1_ng  = (const float*)d_in[1];
  const float* ff1_nb  = (const float*)d_in[2];
  const float* ff1_b1  = (const float*)d_in[4];
  const float* ff1_b2  = (const float*)d_in[6];
  const float* attn_ng = (const float*)d_in[7];
  const float* attn_nb = (const float*)d_in[8];
  const float* kvn_g   = (const float*)d_in[11];
  const float* kvn_b   = (const float*)d_in[12];
  const float* conv_ng = (const float*)d_in[15];
  const float* conv_nb = (const float*)d_in[16];
  const float* pw1_b   = (const float*)d_in[18];
  const float* dw_w    = (const float*)d_in[19];
  const float* dw_b    = (const float*)d_in[20];
  const float* bn_g    = (const float*)d_in[21];
  const float* bn_b    = (const float*)d_in[22];
  const float* bn_rm   = (const float*)d_in[23];
  const float* bn_rv   = (const float*)d_in[24];
  const float* pw2_b   = (const float*)d_in[26];
  const float* ff2_ng  = (const float*)d_in[27];
  const float* ff2_nb  = (const float*)d_in[28];
  const float* ff2_b1  = (const float*)d_in[30];
  const float* ff2_b2  = (const float*)d_in[32];
  const float* fin_g   = (const float*)d_in[33];
  const float* fin_b   = (const float*)d_in[34];

  char* ws = (char*)d_ws;
  bf16_t* Abuf = (bf16_t*)(ws + OFF_A);
  bf16_t* midb = (bf16_t*)(ws + OFF_MID);
  float*  xcur = (float*)(ws + OFF_X);
  float*  qf   = (float*)(ws + OFF_TMP);
  float*  kvaf = qf + 8192ull * 1024;
  float*  kvf  = kvaf + 8192ull * 384;
  float*  pwf  = (float*)(ws + OFF_TMP);
  bf16_t* latb = (bf16_t*)(ws + OFF_LAT);
  bf16_t* qbf  = (bf16_t*)(ws + OFF_Q);
  bf16_t* kbf  = (bf16_t*)(ws + OFF_K);
  bf16_t* vTbf = (bf16_t*)(ws + OFF_VT);
  float*  glub = (float*)(ws + OFF_GLU);
  const bf16_t* w_ff1w1 = (const bf16_t*)(ws + OFF_W0);
  const bf16_t* w_ff1w2 = (const bf16_t*)(ws + OFF_W1);
  const bf16_t* w_wq    = (const bf16_t*)(ws + OFF_W2);
  const bf16_t* w_wkva  = (const bf16_t*)(ws + OFF_W3);
  const bf16_t* w_wkvb  = (const bf16_t*)(ws + OFF_W4);
  const bf16_t* w_wo    = (const bf16_t*)(ws + OFF_W5);
  const bf16_t* w_pw1   = (const bf16_t*)(ws + OFF_W6);
  const bf16_t* w_pw2   = (const bf16_t*)(ws + OFF_W7);
  const bf16_t* w_ff2w1 = (const bf16_t*)(ws + OFF_W8);
  const bf16_t* w_ff2w2 = (const bf16_t*)(ws + OFF_W9);

  // Weight convert+transpose (deterministic per call)
  struct WD { int idx, K, N, Npad; size_t off; };
  const WD wd[10] = {
    {3, 1024, 4096, 4096, OFF_W0}, {5, 4096, 1024, 1024, OFF_W1},
    {9, 1024, 1024, 1024, OFF_W2}, {10, 1024, 320, 384, OFF_W3},
    {13, 256, 512, 512, OFF_W4},   {14, 1024, 1024, 1024, OFF_W5},
    {17, 1024, 2048, 2048, OFF_W6},{25, 1024, 1024, 1024, OFF_W7},
    {29, 1024, 4096, 4096, OFF_W8},{31, 4096, 1024, 1024, OFF_W9},
  };
  for (int i = 0; i < 10; i++) {
    int total = wd[i].Npad * wd[i].K;
    convert_wt<<<dim3((total + 255) / 256), dim3(256), 0, stream>>>(
        (const float*)d_in[wd[i].idx], wd[i].K, wd[i].N, (bf16_t*)(ws + wd[i].off), total);
  }

  // ---- FFN1 (macaron half) ----
  ln_kernel<<<8192, 256, 0, stream>>>(x, 1024, 1024, ff1_ng, ff1_nb, Abuf, nullptr, 1024);
  launch_gemm(2, Abuf, 1024, w_ff1w1, 1024, ff1_b1, nullptr, 0.f, nullptr, midb, 4096, 8192, 4096, 1024, stream);
  launch_gemm(1, midb, 4096, w_ff1w2, 4096, ff1_b2, x, 0.5f, xcur, nullptr, 1024, 8192, 1024, 4096, stream);

  // ---- RMLA attention ----
  ln_kernel<<<8192, 256, 0, stream>>>(xcur, 1024, 1024, attn_ng, attn_nb, Abuf, nullptr, 1024);
  launch_gemm(0, Abuf, 1024, w_wq, 1024, nullptr, nullptr, 0.f, qf, nullptr, 1024, 8192, 1024, 1024, stream);
  launch_gemm(0, Abuf, 1024, w_wkva, 1024, nullptr, nullptr, 0.f, kvaf, nullptr, 384, 8192, 384, 1024, stream);
  ln_kernel<<<8192, 256, 0, stream>>>(kvaf, 384, 256, kvn_g, kvn_b, latb, nullptr, 256);
  launch_gemm(0, latb, 256, w_wkvb, 256, nullptr, nullptr, 0.f, kvf, nullptr, 512, 8192, 512, 256, stream);
  rope_pack<<<(8 * 1024 * 16 * 64) / 256, 256, 0, stream>>>(qf, 1024, 0, 16, qbf, 1, 8 * 1024 * 16 * 64);
  rope_pack<<<(8 * 1024 * 4 * 64) / 256, 256, 0, stream>>>(kvf, 512, 0, 4, kbf, 1, 8 * 1024 * 4 * 64);
  pack_vT<<<(8 * 4 * 64 * 1024) / 256, 256, 0, stream>>>(kvf, vTbf, 8 * 4 * 64 * 1024);
  flash_attn<<<1024, 256, 0, stream>>>(qbf, kbf, vTbf, Abuf);
  launch_gemm(1, Abuf, 1024, w_wo, 1024, nullptr, xcur, 1.f, xcur, nullptr, 1024, 8192, 1024, 1024, stream);

  // ---- Conv module ----
  ln_kernel<<<8192, 256, 0, stream>>>(xcur, 1024, 1024, conv_ng, conv_nb, Abuf, nullptr, 1024);
  launch_gemm(0, Abuf, 1024, w_pw1, 1024, pw1_b, nullptr, 0.f, pwf, nullptr, 2048, 8192, 2048, 1024, stream);
  glu_kernel<<<(8192 * 1024) / 256, 256, 0, stream>>>(pwf, glub, 8192 * 1024);
  dwconv_bn_silu<<<(8192 * 1024) / 256, 256, 0, stream>>>(glub, dw_w, dw_b, bn_g, bn_b, bn_rm, bn_rv, Abuf, 8192 * 1024);
  launch_gemm(1, Abuf, 1024, w_pw2, 1024, pw2_b, xcur, 1.f, xcur, nullptr, 1024, 8192, 1024, 1024, stream);

  // ---- FFN2 (macaron half) ----
  ln_kernel<<<8192, 256, 0, stream>>>(xcur, 1024, 1024, ff2_ng, ff2_nb, Abuf, nullptr, 1024);
  launch_gemm(2, Abuf, 1024, w_ff2w1, 1024, ff2_b1, nullptr, 0.f, nullptr, midb, 4096, 8192, 4096, 1024, stream);
  launch_gemm(1, midb, 4096, w_ff2w2, 4096, ff2_b2, xcur, 0.5f, xcur, nullptr, 1024, 8192, 1024, 4096, stream);

  // ---- Final LN -> d_out (f32) ----
  ln_kernel<<<8192, 256, 0, stream>>>(xcur, 1024, 1024, fin_g, fin_b, nullptr, (float*)d_out, 1024);
}
